// GroupedQueryAttention_52750788329546
// MI455X (gfx1250) — matrix operand order repaired and checked by reading, not compiled, not run on hardware
//
#include <hip/hip_runtime.h>
#include <hip/hip_bf16.h>
#include <stdint.h>

// Problem constants (match reference)
#define B_   2
#define N_   2048
#define DIM_ 1024
#define H_   16
#define HD_  64
#define G_   4
#define GHD_ 256
#define MTOT 4096   // B_*N_

typedef __attribute__((ext_vector_type(16))) __bf16   v16bf;
typedef __attribute__((ext_vector_type(8)))  float    v8f;
typedef __attribute__((ext_vector_type(4)))  unsigned u32x4;

union Frag {
  u32x4 q[2];
  unsigned short s[16];
  v16bf v;
};

// ---- bf16 round-to-nearest-even helpers ----
__device__ __forceinline__ unsigned rnd_bf(float f) {
  unsigned u = __builtin_bit_cast(unsigned, f);
  return u + 0x7FFFu + ((u >> 16) & 1u);
}
__device__ __forceinline__ unsigned short f2bf(float f) {
  return (unsigned short)(rnd_bf(f) >> 16);
}
__device__ __forceinline__ unsigned pack2(float a, float b) {   // {lo=a, hi=b}
#if __has_builtin(__builtin_amdgcn_perm)
  return __builtin_amdgcn_perm(rnd_bf(b), rnd_bf(a), 0x07060302u); // v_perm_b32
#else
  return (rnd_bf(a) >> 16) | (rnd_bf(b) & 0xffff0000u);
#endif
}

// ---- raw v_exp_f32 (no libm range fixups) ----
__device__ __forceinline__ float fexp2(float x) {
#if __has_builtin(__builtin_amdgcn_exp2f)
  return __builtin_amdgcn_exp2f(x);
#else
  return exp2f(x);
#endif
}

// ---- cross-half (lane ^ 16) exchange: v_permlanex16 with identity selects ----
__device__ __forceinline__ unsigned xhalf_u(unsigned v) {
#if __has_builtin(__builtin_amdgcn_permlanex16)
  return (unsigned)__builtin_amdgcn_permlanex16((int)v, (int)v, 0x76543210, 0xfedcba98, false, false);
#else
  return (unsigned)__shfl_xor((int)v, 16);
#endif
}
__device__ __forceinline__ float xhalf_f(float v) {
  return __builtin_bit_cast(float, xhalf_u(__builtin_bit_cast(unsigned, v)));
}

// ---- wave-uniform predicate "am I wave 0 of the block" as a scalar branch ----
// (TDM instructions ignore EXEC, so we must *branch over* them, not mask them.)
__device__ __forceinline__ bool is_wave0() {
  return __builtin_amdgcn_readfirstlane(threadIdx.x >> 5) == 0;
}

// =====================  Tensor Data Mover (CDNA5 TDM)  =====================
#if defined(__has_builtin)
#  if __has_builtin(__builtin_amdgcn_tensor_load_to_lds)
#    define HAVE_TDM 1
#  endif
#endif

typedef __attribute__((ext_vector_type(4))) unsigned tdm_g0_t;
typedef __attribute__((ext_vector_type(8))) int      tdm_g1_t;
typedef __attribute__((ext_vector_type(4))) int      tdm_g2_t;

// 2D tile load Global -> LDS with hardware row padding (bank-conflict-free
// strides). tile0 = elements per row, tile1 = rows, stride0 = tensor row
// pitch (elements), all data_size = 2 bytes. pad encodes D# pad_interval /
// pad_amount (interval: 2<<v DWORDs, amount: v+1 DWORDs).
#ifdef HAVE_TDM
__device__ __forceinline__ void tdm_load_2d(const void* gaddr, const void* lds,
                                            int tile0, int tile1, int stride0,
                                            int pad_interval, int pad_amount) {
  const unsigned long long ga = (unsigned long long)gaddr;
  tdm_g0_t g0;
  g0.x = 1u;                                               // count=1, user mode
  g0.y = (unsigned)(unsigned long long)lds;                // lds_addr (bytes)
  g0.z = (unsigned)ga;                                     // global_addr[31:0]
  g0.w = (unsigned)((ga >> 32) & 0x1FFFFFFu) | (2u << 30); // addr[56:32] | type=2
  tdm_g1_t g1;
  g1[0] = (1 << 16)                    // data_size = 1 (2 bytes)
        | (1 << 20)                    // pad_enable
        | (pad_interval << 22)
        | (pad_amount << 25);
  g1[1] = (tile0 & 0xffff) << 16;                             // tensor_dim0[15:0]
  g1[2] = ((unsigned)tile0 >> 16) | ((tile1 & 0xffff) << 16); // dim0[31:16] | dim1[15:0]
  g1[3] = ((unsigned)tile1 >> 16) | (tile0 << 16);            // dim1[31:16] | tile_dim0
  g1[4] = tile1 & 0xffff;                                     // tile_dim1 (tile_dim2=0)
  g1[5] = stride0;                                            // tensor_dim0_stride[31:0]
  g1[6] = 0;                                                  // stride0 hi | stride1 lo
  g1[7] = 0;
  const tdm_g2_t z4 = {};
#if __clang_major__ >= 23
  const tdm_g1_t z8 = {};
  __builtin_amdgcn_tensor_load_to_lds(g0, g1, z4, z4, z8, 0);
#else
  __builtin_amdgcn_tensor_load_to_lds(g0, g1, z4, z4, 0);
#endif
}
__device__ __forceinline__ void wait_tdm() {
#if __has_builtin(__builtin_amdgcn_s_wait_tensorcnt)
  __builtin_amdgcn_s_wait_tensorcnt(0);
#else
  asm volatile("s_wait_tensorcnt 0x0" ::: "memory");
#endif
}
#endif  // HAVE_TDM

// ---- fallback 16B copy (used only when TDM builtin is unavailable) ----
#define CP16(g, l) (*(u32x4*)(l) = *(const u32x4*)(g))

// ---------------- conversion kernels ----------------
__global__ void cvt_bf16_kernel(const float* __restrict__ in,
                                unsigned short* __restrict__ out, int n) {
  int i = blockIdx.x * blockDim.x + threadIdx.x;
  if (i < n) out[i] = f2bf(in[i]);
}

// in: [K,N] f32 row-major  ->  out: [N,K] bf16 row-major (W^T)
__global__ void cvt_t_bf16_kernel(const float* __restrict__ in,
                                  unsigned short* __restrict__ out, int K, int N) {
  int i = blockIdx.x * blockDim.x + threadIdx.x;
  if (i < K * N) {
    int n = i / K, k = i - n * K;
    out[i] = f2bf(in[(size_t)k * N + n]);
  }
}

// ---------------- GEMM: C = A[M,K] * BT^T + bias ----------------
// All 8 waves of a block share one A panel (nwM*64 rows) staged into
// double-buffered LDS by the TDM (one tensor_load_to_lds per k-step, rows
// hardware-padded 64B+16B -> 80B so ds_load_b128 fragment reads are
// bank-conflict-free).
// MODE 0: bf16 row-major   MODE 1: f32 row-major   MODE 2: bf16 V^T [B,G,HD,N]
template <int MODE>
__global__ void gemm_wmma_kernel(const unsigned short* __restrict__ A,   // [M,K] bf16
                                 const unsigned short* __restrict__ BT,  // [N,K] bf16 (W^T)
                                 const float* __restrict__ bias,         // [N]
                                 void* __restrict__ out,
                                 int M, int N, int K) {
  __shared__ __attribute__((aligned(16))) unsigned short sA[2 * 128 * 40];

  const int t      = threadIdx.x;
  const int lane   = t & 31;
  const int w      = t >> 5;
  const int tilesN = N >> 6;
  const int nwN    = tilesN >= 8 ? 8 : tilesN;  // N-tiles per block
  const int nwM    = 8 / nwN;                   // M-tiles per block
  const int bPerRow = tilesN / nwN;
  const int m0b  = (blockIdx.x / bPerRow) * nwM * 64;
  const int bn   = blockIdx.x % bPerRow;
  const int mt   = w / nwN;
  const int nt   = w % nwN;
  const int n0   = (bn * nwN + nt) * 64;
  const int half = lane >> 4;
  const int lr   = lane & 15;

  auto stage = [&](int k0, int buf) {
#ifdef HAVE_TDM
    if (is_wave0())   // one TDM descriptor covers the whole panel
      tdm_load_2d(A + (size_t)m0b * K + k0, &sA[buf * 5120],
                  /*tile0=*/32, /*tile1=*/nwM * 64, /*stride0=*/K,
                  /*pad_interval=*/3, /*pad_amount=*/3);   // 64B rows + 16B pad
#else
    for (int i = 0; i < nwM; ++i) {
      const int c   = t + (i << 8);
      const int row = c >> 2;
      const int kc8 = (c & 3) << 3;
      CP16(A + (size_t)(m0b + row) * K + k0 + kc8,
           &sA[buf * 5120 + row * 40 + kc8]);
    }
#endif
  };
  auto stage_wait = [&]() {
#ifdef HAVE_TDM
    if (is_wave0()) wait_tdm();
#endif
    __syncthreads();
  };

  v8f acc[4][4] = {};
  stage(0, 0);
  int buf = 0;
  for (int k0 = 0; k0 < K; k0 += 32, buf ^= 1) {
    stage_wait();
    if (k0 + 32 < K) stage(k0 + 32, buf ^ 1);

    Frag af[4], bf[4];
#pragma unroll
    for (int mi = 0; mi < 4; ++mi) {
      // A layout: elem j -> K = k0 + 16*half + j
      const u32x4* lp = (const u32x4*)&sA[buf * 5120 + (mt * 64 + mi * 16 + lr) * 40 + half * 16];
      af[mi].q[0] = lp[0];
      af[mi].q[1] = lp[1];
    }
#pragma unroll
    for (int ni = 0; ni < 4; ++ni) {
      // B layout: elem jj -> K = k0 + 16*half + jj; contiguous in W^T row
      const u32x4* p = (const u32x4*)(BT + (size_t)(n0 + ni * 16 + lr) * K + k0 + half * 16);
      bf[ni].q[0] = p[0];
      bf[ni].q[1] = p[1];
    }
#pragma unroll
    for (int mi = 0; mi < 4; ++mi)
#pragma unroll
      for (int ni = 0; ni < 4; ++ni)
        acc[mi][ni] = __builtin_amdgcn_wmma_f32_16x16x32_bf16(
            false, af[mi].v, false, bf[ni].v, (short)0, acc[mi][ni], false, false);
  }

  const int m0 = m0b + mt * 64;
#pragma unroll
  for (int ni = 0; ni < 4; ++ni) {
    const int   col = n0 + ni * 16 + lr;
    const float bv  = bias[col];
#pragma unroll
    for (int mi = 0; mi < 4; ++mi) {
#pragma unroll
      for (int r = 0; r < 8; ++r) {
        const int   row = m0 + mi * 16 + r + half * 8;   // C layout: M = r + 8*half
        const float val = acc[mi][ni][r] + bv;
        if (MODE == 0) {
          ((unsigned short*)out)[(size_t)row * N + col] = f2bf(val);
        } else if (MODE == 1) {
          ((float*)out)[(size_t)row * N + col] = val;
        } else {  // V^T: [B,G,HD,N]
          const int bb = row >> 11;          // row / N_
          const int n  = row & (N_ - 1);
          const int g  = col >> 6;
          const int d  = col & 63;
          ((unsigned short*)out)[(((size_t)bb * G_ + g) * HD_ + d) * N_ + n] = f2bf(val);
        }
      }
    }
  }
}

// ---------------- flash GQA attention, S^T / O^T formulation ----------------
// All 8 waves of a block share the same (b,h): the TDM stages each 32-key
// K chunk (32x64, rows padded 128B+16B -> 144B) and V^T chunk (64x32, rows
// padded 64B+16B -> 80B) into double-buffered LDS.
// qb: [B,N,H,HD], kb: [B,N,G,HD], vt: [B,G,HD,N], ab: [B,N,H,HD]  (all bf16)
__global__ void gqa_attention_kernel(const unsigned short* __restrict__ qb,
                                     const unsigned short* __restrict__ kb,
                                     const unsigned short* __restrict__ vt,
                                     unsigned short* __restrict__ ab) {
  __shared__ __attribute__((aligned(16))) unsigned short sK[2 * 32 * 72];
  __shared__ __attribute__((aligned(16))) unsigned short sV[2 * 64 * 40];

  const int t0   = threadIdx.x;
  const int lane = t0 & 31;
  const int tw   = blockIdx.x * 8 + (t0 >> 5);
  const int qt   = tw & 127;              // 128 query tiles of 16 per (b,h)
  const int h    = (tw >> 7) & (H_ - 1);
  const int b    = tw >> 11;
  const int g    = h & (G_ - 1);          // head h -> group h % G
  const int q0   = qt << 4;
  const int half = lane >> 4;
  const int lr   = lane & 15;
  const float cexp = 0.125f * 1.44269504f;   // HD^-0.5 * log2(e), folded into exp2

  const unsigned short* kbase = kb + (size_t)b * N_ * GHD_ + g * HD_;
  const unsigned short* vbase = vt + ((size_t)b * G_ + g) * HD_ * (size_t)N_;

  auto stageKV = [&](int kc, int buf) {
#ifdef HAVE_TDM
    if (is_wave0()) {
      tdm_load_2d(kbase + (size_t)kc * GHD_, &sK[buf * 2304],
                  /*tile0=*/64, /*tile1=*/32, /*stride0=*/GHD_,
                  /*pad_interval=*/4, /*pad_amount=*/3);  // 128B rows + 16B pad
      tdm_load_2d(vbase + kc, &sV[buf * 2560],
                  /*tile0=*/32, /*tile1=*/64, /*stride0=*/N_,
                  /*pad_interval=*/3, /*pad_amount=*/3);  // 64B rows + 16B pad
    }
#else
    {  // K chunk: 32 rows(keys) x 64 elems, one 16B chunk per thread
      const int row = t0 >> 3, c8 = (t0 & 7) << 3;
      CP16(kbase + (size_t)(kc + row) * GHD_ + c8, &sK[buf * 2304 + row * 72 + c8]);
    }
    {  // V^T chunk: 64 rows(d) x 32 elems, one 16B chunk per thread
      const int row = t0 >> 2, c8 = (t0 & 3) << 3;
      CP16(vbase + (size_t)row * N_ + kc + c8, &sV[buf * 2560 + row * 40 + c8]);
    }
#endif
  };
  auto stage_wait = [&]() {
#ifdef HAVE_TDM
    if (is_wave0()) wait_tdm();
#endif
    __syncthreads();
  };

  // Q^T B-fragments (loop invariant): elem jj -> d = ds*32 + 16*half + jj, q = lr
  Frag bq[2];
#pragma unroll
  for (int ds = 0; ds < 2; ++ds) {
    const u32x4* p = (const u32x4*)(qb + (size_t)(b * N_ + q0 + lr) * DIM_ + h * HD_ + ds * 32 + half * 16);
    bq[ds].q[0] = p[0];
    bq[ds].q[1] = p[1];
  }

  v8f   o[4] = {};                        // O^T tiles: d = dt*16 + r + 8*half, q = lr
  float mrun = -1e30f, lsum = 0.0f;
  const v8f zero = {};

  stageKV(0, 0);
  int buf = 0;
  for (int kc = 0; kc < N_; kc += 32, buf ^= 1) {
    stage_wait();
    if (kc + 32 < N_) stageKV(kc + 32, buf ^ 1);

    // ---- S^T = K_chunk (keys x d) * Q^T (d x q), two 16-key tiles ----
    v8f s[2];
#pragma unroll
    for (int tt = 0; tt < 2; ++tt) {
      Frag ak0, ak1;
      const unsigned short* krow = &sK[buf * 2304 + (tt * 16 + lr) * 72];
      const u32x4* p0 = (const u32x4*)(krow + half * 16);        // d = 0..31
      ak0.q[0] = p0[0]; ak0.q[1] = p0[1];
      const u32x4* p1 = (const u32x4*)(krow + 32 + half * 16);   // d = 32..63
      ak1.q[0] = p1[0]; ak1.q[1] = p1[1];
      v8f st = __builtin_amdgcn_wmma_f32_16x16x32_bf16(false, ak0.v, false, bq[0].v, (short)0, zero, false, false);
      st     = __builtin_amdgcn_wmma_f32_16x16x32_bf16(false, ak1.v, false, bq[1].v, (short)0, st,   false, false);
      s[tt] = st;
    }

    // ---- online softmax on raw scores; scale folded into exp2 ----
    float mloc = -1e30f;
#pragma unroll
    for (int tt = 0; tt < 2; ++tt)
#pragma unroll
      for (int r = 0; r < 8; ++r) mloc = fmaxf(mloc, s[tt][r]);
    mloc = fmaxf(mloc, xhalf_f(mloc));
    const float mnew  = fmaxf(mrun, mloc);
    const float alpha = fexp2((mrun - mnew) * cexp);
#pragma unroll
    for (int i = 0; i < 4; ++i)
#pragma unroll
      for (int r = 0; r < 8; ++r) o[i][r] *= alpha;
    lsum *= alpha;

    float rs = 0.0f;
#pragma unroll
    for (int tt = 0; tt < 2; ++tt)
#pragma unroll
      for (int r = 0; r < 8; ++r) {
        const float p = fexp2((s[tt][r] - mnew) * cexp);
        s[tt][r] = p;
        rs += p;
      }
    rs += xhalf_f(rs);
    lsum += rs;
    mrun = mnew;

    // ---- P^T (C layout) -> B fragment: pack bf16 pair, one exchange per VGPR ----
    Frag pf;
#pragma unroll
    for (int p = 0; p < 8; ++p) {
      const unsigned loc = pack2(s[0][p], s[1][p]);   // lo = C0 tile, hi = C1 tile
      const unsigned rem = xhalf_u(loc);
      pf.s[p]     = (unsigned short)(half ? (rem >> 16) : (loc & 0xffffu));  // jj = p
      pf.s[p + 8] = (unsigned short)(half ? (loc >> 16) : (rem & 0xffffu));  // jj = p+8
    }

    // ---- O^T += V^T (d x keys) * P^T (keys x q) ----
#pragma unroll
    for (int dt = 0; dt < 4; ++dt) {
      Frag av;
      const u32x4* p = (const u32x4*)&sV[buf * 2560 + (dt * 16 + lr) * 40 + half * 16];
      av.q[0] = p[0]; av.q[1] = p[1];
      o[dt] = __builtin_amdgcn_wmma_f32_16x16x32_bf16(false, av.v, false, pf.v, (short)0, o[dt], false, false);
    }
  }

  // ---- normalize and store: 8 contiguous d per tile -> packed 16B store ----
  const float  inv  = 1.0f / lsum;
  const size_t orow = (size_t)(b * N_ + q0 + lr) * DIM_ + h * HD_;
#pragma unroll
  for (int dt = 0; dt < 4; ++dt) {
    u32x4 wv;
    wv.x = pack2(o[dt][0] * inv, o[dt][1] * inv);
    wv.y = pack2(o[dt][2] * inv, o[dt][3] * inv);
    wv.z = pack2(o[dt][4] * inv, o[dt][5] * inv);
    wv.w = pack2(o[dt][6] * inv, o[dt][7] * inv);
    *(u32x4*)(ab + orow + dt * 16 + half * 8) = wv;
  }
}

extern "C" void kernel_launch(void* const* d_in, const int* in_sizes, int n_in,
                              void* d_out, int out_size, void* d_ws, size_t ws_size,
                              hipStream_t stream) {
  const float* x  = (const float*)d_in[0];
  const float* Wq = (const float*)d_in[1];
  const float* bq = (const float*)d_in[2];
  const float* Wk = (const float*)d_in[3];
  const float* bk = (const float*)d_in[4];
  const float* Wv = (const float*)d_in[5];
  const float* bv = (const float*)d_in[6];
  const float* Wo = (const float*)d_in[7];
  const float* bo = (const float*)d_in[8];
  float* out = (float*)d_out;

  char* ws = (char*)d_ws;
  unsigned short* xb  = (unsigned short*)ws; ws += (size_t)MTOT * DIM_ * 2;
  unsigned short* wqT = (unsigned short*)ws; ws += (size_t)DIM_ * DIM_ * 2;
  unsigned short* wkT = (unsigned short*)ws; ws += (size_t)GHD_ * DIM_ * 2;
  unsigned short* wvT = (unsigned short*)ws; ws += (size_t)GHD_ * DIM_ * 2;
  unsigned short* woT = (unsigned short*)ws; ws += (size_t)DIM_ * DIM_ * 2;
  unsigned short* qb  = (unsigned short*)ws; ws += (size_t)MTOT * DIM_ * 2;
  unsigned short* kb  = (unsigned short*)ws; ws += (size_t)MTOT * GHD_ * 2;
  unsigned short* vt  = (unsigned short*)ws; ws += (size_t)MTOT * GHD_ * 2;
  unsigned short* ab  = (unsigned short*)ws; ws += (size_t)MTOT * DIM_ * 2;

  const int TPB = 256;
  // fp32 -> bf16 (+ weight transpose)
  cvt_bf16_kernel<<<(MTOT * DIM_) / TPB, TPB, 0, stream>>>(x, xb, MTOT * DIM_);
  cvt_t_bf16_kernel<<<(DIM_ * DIM_) / TPB, TPB, 0, stream>>>(Wq, wqT, DIM_, DIM_);
  cvt_t_bf16_kernel<<<(DIM_ * GHD_) / TPB, TPB, 0, stream>>>(Wk, wkT, DIM_, GHD_);
  cvt_t_bf16_kernel<<<(DIM_ * GHD_) / TPB, TPB, 0, stream>>>(Wv, wvT, DIM_, GHD_);
  cvt_t_bf16_kernel<<<(DIM_ * DIM_) / TPB, TPB, 0, stream>>>(Wo, woT, DIM_, DIM_);
  // projections  (grid = (tilesM/nwM) * (tilesN/nwN) blocks, 8 waves each)
  gemm_wmma_kernel<0><<<(MTOT / 64) * (DIM_ / 64) / 8, TPB, 0, stream>>>(xb, wqT, bq, qb, MTOT, DIM_, DIM_);
  gemm_wmma_kernel<0><<<(MTOT / 64) * (GHD_ / 64) / 8, TPB, 0, stream>>>(xb, wkT, bk, kb, MTOT, GHD_, DIM_);
  gemm_wmma_kernel<2><<<(MTOT / 64) * (GHD_ / 64) / 8, TPB, 0, stream>>>(xb, wvT, bv, vt, MTOT, GHD_, DIM_);
  // attention: one wave per 16 query rows, 8 waves (same b,h) per block
  gqa_attention_kernel<<<(B_ * H_ * (N_ / 16)) / 8, TPB, 0, stream>>>(qb, kb, vt, ab);
  // output projection (f32 + bias into d_out)
  gemm_wmma_kernel<1><<<(MTOT / 64) * (DIM_ / 64) / 8, TPB, 0, stream>>>(ab, woT, bo, out, MTOT, DIM_, DIM_);
}
